// GCN_57930518888455
// MI455X (gfx1250) — matrix operand order repaired and checked by reading, not compiled, not run on hardware
//
#include <hip/hip_runtime.h>
#include <hip/hip_bf16.h>

// ---------- WMMA vector types (CDNA5 / gfx1250, wave32) ----------
typedef __attribute__((ext_vector_type(16))) __bf16 v16bf;
typedef __attribute__((ext_vector_type(8)))  __bf16 v8bf;
typedef __attribute__((ext_vector_type(8)))  float  v8f;

// ================= elementwise helpers =================

__global__ void k_cast_bf16(const float* __restrict__ x, __bf16* __restrict__ y, long long n) {
    long long i = (long long)blockIdx.x * blockDim.x + threadIdx.x;
    if (i < n) y[i] = (__bf16)x[i];
}

// W [K x F] f32 row-major -> Wt [F x K] bf16 row-major (B^T, so B-fragments are contiguous)
__global__ void k_transpose_cast(const float* __restrict__ W, __bf16* __restrict__ Wt, int K, int F) {
    int i = blockIdx.x * blockDim.x + threadIdx.x;
    if (i >= K * F) return;
    int k = i / F, f = i % F;
    Wt[(size_t)f * K + k] = (__bf16)W[i];
}

__global__ void k_deg_init(float* __restrict__ deg, int n) {
    int i = blockIdx.x * blockDim.x + threadIdx.x;
    if (i < n) deg[i] = 1.0f;                 // self-loop contribution
}

__global__ void k_deg_edges(const long long* __restrict__ dst, float* __restrict__ deg, long long E) {
    long long e = (long long)blockIdx.x * blockDim.x + threadIdx.x;
    if (e < E) atomicAdd(&deg[(int)dst[e]], 1.0f);
}

__global__ void k_deg_to_dis(float* __restrict__ deg, int n) {
    int i = blockIdx.x * blockDim.x + threadIdx.x;
    if (i < n) {
        float d = deg[i];
        deg[i] = (d > 0.0f) ? rsqrtf(d) : 0.0f;
    }
}

// ================= bf16 WMMA GEMM: C[M x F] = A[M x K] * B[K x F] =================
// A: bf16 row-major.  Bt: bf16 [F x K] row-major (= B transposed).  C: f32.
// Block = 256 threads = 8 waves. Wave tile = 16 rows x 64 cols (4 x 16x16 WMMA
// accumulators) so each A fragment feeds 4 WMMAs: 2 A-loads + 8 B-loads per
// 4 WMMAs. Block tile = 128 rows x 64 cols.
__global__ void k_gemm_bf16_wmma(const __bf16* __restrict__ A,
                                 const __bf16* __restrict__ Bt,
                                 float* __restrict__ C,
                                 int M, int K, int F) {
    const int wave = threadIdx.x >> 5;
    const int lane = threadIdx.x & 31;
    const int l15  = lane & 15;
    const int hi   = lane >> 4;

    const int row0 = (blockIdx.x * 8 + wave) * 16;          // 16-row tile base
    const int col0 = blockIdx.y * 64;                       // 64-col group base
    if (row0 >= M) return;

    int arow = row0 + l15;
    if (arow >= M) arow = M - 1;                            // clamp loads, guard stores

    const __bf16* __restrict__ Ar = A  + (size_t)arow * K;
    // B tile t covers cols [col0 + 16t, col0 + 16t + 16); lane supplies column col0+16t+l15
    const __bf16* __restrict__ Br = Bt + (size_t)(col0 + l15) * K;
    const size_t bstep = (size_t)16 * K;                    // advance 16 columns of B^T

    v8f acc0 = {}, acc1 = {}, acc2 = {}, acc3 = {};
    for (int kb = 0; kb < K; kb += 32) {
        __builtin_prefetch(Ar + kb + 64, 0, 1);             // global_prefetch_b8 hint
        // A fragment: VGPR v<4 -> K = hi*8 + i (i=0..7); v>=4 -> K = 16 + hi*8 + i
        v8bf a0 = *(const v8bf*)(Ar + kb + hi * 8);
        v8bf a1 = *(const v8bf*)(Ar + kb + 16 + hi * 8);
        v16bf af;
#pragma unroll
        for (int i = 0; i < 8; ++i) { af[i] = a0[i]; af[8 + i] = a1[i]; }

        // B fragment per tile: lane = column, i<8 -> K = hi*8 + i; i>=8 -> K = 16 + hi*8 + (i-8)
#pragma unroll
        for (int t = 0; t < 4; ++t) {
            const __bf16* bp = Br + (size_t)t * bstep + kb + hi * 8;
            v8bf b0 = *(const v8bf*)(bp);
            v8bf b1 = *(const v8bf*)(bp + 16);
            v16bf bf;
#pragma unroll
            for (int i = 0; i < 8; ++i) { bf[i] = b0[i]; bf[8 + i] = b1[i]; }
            v8f* acc = (t == 0) ? &acc0 : (t == 1) ? &acc1 : (t == 2) ? &acc2 : &acc3;
            *acc = __builtin_amdgcn_wmma_f32_16x16x32_bf16(
                /*neg_a=*/false, af, /*neg_b=*/false, bf,
                /*c_mod=*/(short)0, *acc, /*reuse_a=*/false, /*reuse_b=*/false);
        }
    }
    // C layout: VGPR j -> row = row0 + j + hi*8, col = col0 + 16t + l15
#pragma unroll
    for (int t = 0; t < 4; ++t) {
        const v8f& acc = (t == 0) ? acc0 : (t == 1) ? acc1 : (t == 2) ? acc2 : acc3;
#pragma unroll
        for (int j = 0; j < 8; ++j) {
            const int r = row0 + j + hi * 8;
            if (r < M) C[(size_t)r * F + col0 + t * 16 + l15] = acc[j];
        }
    }
}

// ================= aggregation =================

// OUT[i,:] = dis[i]^2 * H[i,:]   (self-loop term; also initializes OUT)
__global__ void k_agg_self(const float* __restrict__ H, const float* __restrict__ dis,
                           float* __restrict__ OUT, long long total, int F) {
    long long i = (long long)blockIdx.x * blockDim.x + threadIdx.x;
    if (i >= total) return;
    const float d = dis[(int)(i / F)];
    OUT[i] = d * d * H[i];
}

// OUT[dst,:] += dis[src]*dis[dst] * H[src,:] ; one thread per (edge, 4-float chunk)
__global__ void k_agg_edges(const long long* __restrict__ src, const long long* __restrict__ dst,
                            const float* __restrict__ dis, const float* __restrict__ H,
                            float* __restrict__ OUT, long long E, int F) {
    const int chunks = F >> 2;
    long long gid = (long long)blockIdx.x * blockDim.x + threadIdx.x;
    long long e = gid / chunks;
    if (e >= E) return;
    const int c = (int)(gid % chunks);
    const int s = (int)src[e], d = (int)dst[e];
    const float nrm = dis[s] * dis[d];
    const float4 h = ((const float4*)(H + (size_t)s * F))[c];
    float* o = OUT + (size_t)d * F + (size_t)c * 4;
    atomicAdd(o + 0, nrm * h.x);
    atomicAdd(o + 1, nrm * h.y);
    atomicAdd(o + 2, nrm * h.z);
    atomicAdd(o + 3, nrm * h.w);
}

// y = bf16(relu(x + b[col]))  — feeds next layer's GEMM
__global__ void k_bias_relu_cast(const float* __restrict__ X, const float* __restrict__ b,
                                 __bf16* __restrict__ Y, long long total, int F) {
    long long i = (long long)blockIdx.x * blockDim.x + threadIdx.x;
    if (i >= total) return;
    float v = X[i] + b[(int)(i % F)];
    Y[i] = (__bf16)fmaxf(v, 0.0f);
}

// out[row,:] = log_softmax(X[row,:] + b) over 64 cols; one wave32 per row, 2 cols/lane
__global__ void k_logsoftmax64(const float* __restrict__ X, const float* __restrict__ b,
                               float* __restrict__ out, int N) {
    const int wave = threadIdx.x >> 5;
    const int lane = threadIdx.x & 31;
    const int row = blockIdx.x * 8 + wave;
    if (row >= N) return;
    const float* xr = X + (size_t)row * 64;
    float v0 = xr[lane]      + b[lane];
    float v1 = xr[lane + 32] + b[lane + 32];
    float m = fmaxf(v0, v1);
#pragma unroll
    for (int off = 16; off > 0; off >>= 1) m = fmaxf(m, __shfl_xor(m, off, 32));
    float s = __expf(v0 - m) + __expf(v1 - m);
#pragma unroll
    for (int off = 16; off > 0; off >>= 1) s += __shfl_xor(s, off, 32);
    const float lse = m + __logf(s);
    out[(size_t)row * 64 + lane]      = v0 - lse;
    out[(size_t)row * 64 + lane + 32] = v1 - lse;
}

// ================= host driver =================

static inline long long cdiv(long long a, long long b) { return (a + b - 1) / b; }

extern "C" void kernel_launch(void* const* d_in, const int* in_sizes, int n_in,
                              void* d_out, int out_size, void* d_ws, size_t ws_size,
                              hipStream_t stream) {
    const float*     feats = (const float*)d_in[0];
    const long long* adj   = (const long long*)d_in[1];
    const float*     W1 = (const float*)d_in[2];
    const float*     b1 = (const float*)d_in[3];
    const float*     W2 = (const float*)d_in[4];
    const float*     b2 = (const float*)d_in[5];
    const float*     W3 = (const float*)d_in[6];
    const float*     b3 = (const float*)d_in[7];

    const int HID = in_sizes[3];                 // 128
    const int OUT = in_sizes[7];                 // 64
    const int IN  = in_sizes[2] / HID;           // 512
    const int N   = in_sizes[0] / IN;            // 50000
    const long long E = in_sizes[1] / 2;         // 800000
    const long long* src = adj;
    const long long* dst = adj + E;

    // ---- carve workspace (256B aligned) ----
    char* ws = (char*)d_ws;
    size_t off = 0;
    auto carve = [&](size_t bytes) -> char* {
        char* p = ws + off;
        off += (bytes + 255) & ~(size_t)255;
        return p;
    };
    __bf16* Xbf = (__bf16*)carve((size_t)N * IN * 2);
    __bf16* Hbf = (__bf16*)carve((size_t)N * HID * 2);
    float*  Hf  = (float*) carve((size_t)N * HID * 4);
    float*  AGG = (float*) carve((size_t)N * HID * 4);
    __bf16* W1t = (__bf16*)carve((size_t)IN  * HID * 2);
    __bf16* W2t = (__bf16*)carve((size_t)HID * HID * 2);
    __bf16* W3t = (__bf16*)carve((size_t)HID * OUT * 2);
    float*  dis = (float*) carve((size_t)N * 4);
    (void)ws_size;

    const int T = 256;

    // ---- precision conversion + weight transpose ----
    {
        long long n = (long long)N * IN;
        k_cast_bf16<<<(unsigned)cdiv(n, T), T, 0, stream>>>(feats, Xbf, n);
        k_transpose_cast<<<(unsigned)cdiv((long long)IN  * HID, T), T, 0, stream>>>(W1, W1t, IN,  HID);
        k_transpose_cast<<<(unsigned)cdiv((long long)HID * HID, T), T, 0, stream>>>(W2, W2t, HID, HID);
        k_transpose_cast<<<(unsigned)cdiv((long long)HID * OUT, T), T, 0, stream>>>(W3, W3t, HID, OUT);
    }

    // ---- symmetric normalization: dis = rsqrt(1 + indeg) ----
    k_deg_init<<<(unsigned)cdiv(N, T), T, 0, stream>>>(dis, N);
    k_deg_edges<<<(unsigned)cdiv(E, T), T, 0, stream>>>(dst, dis, E);
    k_deg_to_dis<<<(unsigned)cdiv(N, T), T, 0, stream>>>(dis, N);

    const unsigned mblocks = (unsigned)cdiv(N, 128);   // 8 waves x 16 rows per block

    // ---- layer 1: [N,IN] @ [IN,HID] ----
    {
        dim3 grid(mblocks, (unsigned)(HID / 64));
        k_gemm_bf16_wmma<<<grid, T, 0, stream>>>(Xbf, W1t, Hf, N, IN, HID);
        long long tot = (long long)N * HID;
        k_agg_self<<<(unsigned)cdiv(tot, T), T, 0, stream>>>(Hf, dis, AGG, tot, HID);
        k_agg_edges<<<(unsigned)cdiv(E * (HID / 4), T), T, 0, stream>>>(src, dst, dis, Hf, AGG, E, HID);
        k_bias_relu_cast<<<(unsigned)cdiv(tot, T), T, 0, stream>>>(AGG, b1, Hbf, tot, HID);
    }
    // ---- layer 2: [N,HID] @ [HID,HID] ----
    {
        dim3 grid(mblocks, (unsigned)(HID / 64));
        k_gemm_bf16_wmma<<<grid, T, 0, stream>>>(Hbf, W2t, Hf, N, HID, HID);
        long long tot = (long long)N * HID;
        k_agg_self<<<(unsigned)cdiv(tot, T), T, 0, stream>>>(Hf, dis, AGG, tot, HID);
        k_agg_edges<<<(unsigned)cdiv(E * (HID / 4), T), T, 0, stream>>>(src, dst, dis, Hf, AGG, E, HID);
        k_bias_relu_cast<<<(unsigned)cdiv(tot, T), T, 0, stream>>>(AGG, b2, Hbf, tot, HID);
    }
    // ---- layer 3: [N,HID] @ [HID,OUT] + bias + log_softmax ----
    {
        dim3 grid(mblocks, (unsigned)(OUT / 64));
        k_gemm_bf16_wmma<<<grid, T, 0, stream>>>(Hbf, W3t, Hf, N, HID, OUT);
        long long tot = (long long)N * OUT;
        k_agg_self<<<(unsigned)cdiv(tot, T), T, 0, stream>>>(Hf, dis, AGG, tot, OUT);
        k_agg_edges<<<(unsigned)cdiv(E * (OUT / 4), T), T, 0, stream>>>(src, dst, dis, Hf, AGG, E, OUT);
        k_logsoftmax64<<<(unsigned)cdiv(N, 8), T, 0, stream>>>(AGG, b3, (float*)d_out, N);
    }
}
